// LocalAttention_19035295056282
// MI455X (gfx1250) — matrix operand order repaired and checked by reading, not compiled, not run on hardware
//
#include <hip/hip_runtime.h>
#include <hip/hip_bf16.h>

typedef _Float16 h4  __attribute__((ext_vector_type(4)));
typedef _Float16 h8  __attribute__((ext_vector_type(8)));
typedef _Float16 h16 __attribute__((ext_vector_type(16)));
typedef float    f4  __attribute__((ext_vector_type(4)));
typedef float    f8  __attribute__((ext_vector_type(8)));

#define WMMA_F16(A, B, C) \
    __builtin_amdgcn_wmma_f32_16x16x32_f16(false, (A), false, (B), (short)0, (C), false, false)

__device__ __forceinline__ h16 cat8(h8 lo, h8 hi) {
    return __builtin_shufflevector(lo, hi, 0,1,2,3,4,5,6,7,8,9,10,11,12,13,14,15);
}

constexpr int B_ = 2;
constexpr int E_ = 256;   // q/k/v output channels (= H*D)
constexpr int T_ = 2048;
constexpr int C_ = 2048;  // input channels E*F
constexpr int H_ = 8;
constexpr int P_ = 4;
constexpr int D_ = 32;    // head dim

// ---------------------------------------------------------------------------
// f32 -> f16 weight conversion
// ---------------------------------------------------------------------------
__global__ void cvt_f16_kernel(const float* __restrict__ src,
                               _Float16* __restrict__ dst, int n) {
    int i = blockIdx.x * blockDim.x + threadIdx.x;
    if (i < n) dst[i] = (_Float16)src[i];
}

// ---------------------------------------------------------------------------
// Pointwise-conv GEMM:  y[b][t][o] = sum_c W[o][c] * x[b][c][t] + bias[o]
// A = x^T tile [16t x 32c] (LDS-staged, transposed), B = W^T [32c x 16o]
// C-tile: lane = o column, VGPR r = t row (r, or r+8 for lanes 16-31).
// OT = number of 16-wide o-tiles per pass (4 => 64 outputs / wg / z-slice)
// ---------------------------------------------------------------------------
template<int OT>
__global__ __launch_bounds__(256)
void proj_gemm_kernel(const float* __restrict__ x,
                      const _Float16* __restrict__ Wh,
                      const float* __restrict__ bias,
                      _Float16* __restrict__ y, int O) {
    __shared__ _Float16 xT[128 * 40];   // [t_local][c_local], stride 40 (80B, 16B-aligned, bank-spread)

    const int tb   = blockIdx.x * 128;
    const int b    = blockIdx.y;
    const int o0   = blockIdx.z * (OT * 16);
    const int tid  = threadIdx.x;
    const int wave = tid >> 5;
    const int lane = tid & 31;
    const int ln   = lane & 15;
    const bool hib = lane >= 16;

    f8 acc[OT];
#pragma unroll
    for (int ti = 0; ti < OT; ++ti) acc[ti] = (f8){};
    float bv[OT];
#pragma unroll
    for (int ti = 0; ti < OT; ++ti) bv[ti] = bias[o0 + ti * 16 + ln];

    for (int c0 = 0; c0 < C_; c0 += 32) {
        __syncthreads();
        {   // cooperatively stage x^T: 32 c x 128 t, converting f32 -> f16
            const int c  = tid >> 3;         // 0..31
            const int ts = (tid & 7) * 16;   // 0..112
            const float* xp = x + ((size_t)b * C_ + c0 + c) * T_ + tb + ts;
#pragma unroll
            for (int i = 0; i < 16; ++i)
                xT[(ts + i) * 40 + c] = (_Float16)xp[i];
        }
        __syncthreads();

        // A operand: row t = lane%16, K-pairs: lanes0-15 c{0..7,16..23}, hi c{8..15,24..31}
        const int row = wave * 16 + ln;
        h8 alo = *(const h8*)&xT[row * 40 + (hib ? 8 : 0)];
        h8 ahi = *(const h8*)&xT[row * 40 + (hib ? 24 : 16)];
        h16 A = cat8(alo, ahi);

#pragma unroll
        for (int ti = 0; ti < OT; ++ti) {
            // B operand: N = o = lane%16, K = c: lanes0-15 c{0..7,16..23}, hi c{8..15,24..31} (row of W)
            const _Float16* wp = Wh + (size_t)(o0 + ti * 16 + ln) * C_ + c0 + (hib ? 8 : 0);
            h16 Bm = cat8(*(const h8*)wp, *(const h8*)(wp + 16));
            acc[ti] = WMMA_F16(A, Bm, acc[ti]);
        }
    }

#pragma unroll
    for (int ti = 0; ti < OT; ++ti) {
#pragma unroll
        for (int r = 0; r < 8; ++r) {
            const int t = tb + wave * 16 + r + (hib ? 8 : 0);
            y[((size_t)b * T_ + t) * O + o0 + ti * 16 + ln] =
                (_Float16)(acc[ti][r] + bv[ti]);
        }
    }
}

// ---------------------------------------------------------------------------
// slope[b][h][t] = sum_p (p+1) * sigmoid(pbuf[b][t][h*P+p])
// ---------------------------------------------------------------------------
__global__ void slope_kernel(const _Float16* __restrict__ pbuf,
                             float* __restrict__ slope) {
    const int i = blockIdx.x * blockDim.x + threadIdx.x;  // b*T + t
    const int b = i / T_, t = i % T_;
    const _Float16* pp = pbuf + (size_t)i * (H_ * P_);
#pragma unroll
    for (int h = 0; h < H_; ++h) {
        float s = 0.f;
#pragma unroll
        for (int p = 0; p < P_; ++p) {
            float v  = (float)pp[h * P_ + p];
            float sg = 1.f / (1.f + __expf(-v));
            s += (float)(p + 1) * sg;
        }
        slope[((size_t)b * H_ + h) * T_ + t] = s;
    }
}

// ---------------------------------------------------------------------------
// Fused flash attention.  Computes S^T = K Q^T per 16x16 tile so that the
// softmax runs in-lane and the probability C-tile is already in WMMA
// A-operand layout for the PV product.  Penalty: slope[t]*|t-s|.
// Workgroup: 8 waves = 128 t rows of one (b,h); loop s over T in steps of 32.
// ---------------------------------------------------------------------------
__global__ __launch_bounds__(256)
void attn_kernel(const _Float16* __restrict__ q,
                 const _Float16* __restrict__ k,
                 const _Float16* __restrict__ v,
                 const float* __restrict__ slope,
                 float* __restrict__ attnT) {
    __shared__ _Float16 vT[32 * 40];   // V^T tile [d][s], stride 40 halves

    const int tb   = blockIdx.x * 128;
    const int bh   = blockIdx.y;
    const int b    = bh >> 3, h = bh & 7;
    const int tid  = threadIdx.x;
    const int wave = tid >> 5;
    const int lane = tid & 31;
    const int ln   = lane & 15;
    const bool hib = lane >= 16;
    const int tw   = tb + wave * 16;

    // Q as B operand [32d x 16t]: lane = t, lanes0-15 d{0..7,16..23}, hi d{8..15,24..31}
    const _Float16* qp = q + ((size_t)b * T_ + tw + ln) * E_ + h * D_ + (hib ? 8 : 0);
    const h16 QB = cat8(*(const h8*)qp, *(const h8*)(qp + 16));

    const float slp = slope[(size_t)bh * T_ + tw + ln];
    const float tA  = (float)(tw + ln);

    float m = -3.0e38f, l = 0.f;
    f8 acc0 = (f8){}, acc1 = (f8){};

    for (int s0 = 0; s0 < T_; s0 += 32) {
        __syncthreads();
        {   // cooperative V^T staging: 32 s x 32 d  ->  vT[d][s]
            const int s  = tid >> 3;
            const int d0 = (tid & 7) * 4;
            const _Float16* vp = v + ((size_t)b * T_ + s0 + s) * E_ + h * D_ + d0;
            h4 vv = *(const h4*)vp;
#pragma unroll
            for (int j = 0; j < 4; ++j) vT[(d0 + j) * 40 + s] = vv[j];
        }
        __syncthreads();

        // K rows as A operands [16s x 32d]; lanes0-15 d{0..7,16..23}, hi d{8..15,24..31}
        const _Float16* kp = k + ((size_t)b * T_ + s0 + ln) * E_ + h * D_ + (hib ? 8 : 0);
        h16 A0 = cat8(*(const h8*)kp, *(const h8*)(kp + 16));
        const _Float16* kp1 = kp + (size_t)16 * E_;
        h16 A1 = cat8(*(const h8*)kp1, *(const h8*)(kp1 + 16));
        __builtin_prefetch(kp + (size_t)32 * E_, 0, 1);  // next s-block

        f8 z = (f8){};
        f8 st0 = WMMA_F16(A0, QB, z);   // S^T rows s0..s0+15
        f8 st1 = WMMA_F16(A1, QB, z);   // S^T rows s0+16..s0+31

        // scores + distance penalty; online softmax stats (in-lane + xor16)
        float sc0[8], sc1[8];
        float mloc = -3.0e38f;
#pragma unroll
        for (int r = 0; r < 8; ++r) {
            const float sA = (float)(s0 + r + (hib ? 8 : 0));
            sc0[r] = st0[r] - slp * fabsf(tA - sA);
            sc1[r] = st1[r] - slp * fabsf(tA - (sA + 16.f));
            mloc = fmaxf(mloc, fmaxf(sc0[r], sc1[r]));
        }
        mloc = fmaxf(mloc, __shfl_xor(mloc, 16));
        const float mnew  = fmaxf(m, mloc);
        const float scale = __expf(m - mnew);

        float rsum = 0.f;
        h8 p0, p1;
#pragma unroll
        for (int r = 0; r < 8; ++r) {
            const float e0 = __expf(sc0[r] - mnew);
            const float e1 = __expf(sc1[r] - mnew);
            rsum += e0 + e1;
            p0[r] = (_Float16)e0;
            p1[r] = (_Float16)e1;
        }
        rsum += __shfl_xor(rsum, 16);
        l = l * scale + rsum;
        m = mnew;

        // rescale output accumulators by row factor (row = t)
#pragma unroll
        for (int r = 0; r < 8; ++r) {
            const float f = __shfl(scale, r + (hib ? 8 : 0));
            acc0[r] *= f;
            acc1[r] *= f;
        }

        // Probability C-tile is already A-layout: K pairs {r},{16+r} in-lane.
        h16 AP = cat8(p0, p1);

        // V as B operands [32s x 16d] from vT: lane = d, lanes0-15 s{0..7,16..23}, hi s{8..15,24..31}
        const int off = hib ? 8 : 0;
        h16 BV0 = cat8(*(const h8*)&vT[ln * 40 + off],
                       *(const h8*)&vT[ln * 40 + off + 16]);
        h16 BV1 = cat8(*(const h8*)&vT[(16 + ln) * 40 + off],
                       *(const h8*)&vT[(16 + ln) * 40 + off + 16]);

        acc0 = WMMA_F16(AP, BV0, acc0);   // out cols d 0..15
        acc1 = WMMA_F16(AP, BV1, acc1);   // out cols d 16..31
    }

    const float linv = 1.f / l;
#pragma unroll
    for (int r = 0; r < 8; ++r) {
        const float f = __shfl(linv, r + (hib ? 8 : 0));
        const int t = tw + r + (hib ? 8 : 0);
        float* op = attnT + ((size_t)b * T_ + t) * E_ + h * D_;
        op[ln]      = acc0[r] * f;
        op[16 + ln] = acc1[r] * f;
    }
}

// ---------------------------------------------------------------------------
// Output projection: out[b][co][t] = sum_e Wo[co][e] * attnT[b][t][e] + bo[co]
// A = Wo tile [16co x 32e], B = attn [32e x 16t] (cvt f32->f16 on load).
// C-tile: lane = t column -> coalesced f32 stores along t.
// ---------------------------------------------------------------------------
__global__ __launch_bounds__(256)
void out_proj_kernel(const float* __restrict__ attnT,
                     const _Float16* __restrict__ Woh,
                     const float* __restrict__ bo,
                     float* __restrict__ out) {
    const int tb   = blockIdx.x * 128;
    const int b    = blockIdx.y;
    const int cob  = blockIdx.z * 256;
    const int tid  = threadIdx.x;
    const int wave = tid >> 5;
    const int lane = tid & 31;
    const int ln   = lane & 15;
    const bool hib = lane >= 16;
    const int t    = tb + wave * 16 + ln;

    for (int pass = 0; pass < 4; ++pass) {
        const int co0 = cob + pass * 64;
        f8 acc[4];
#pragma unroll
        for (int ti = 0; ti < 4; ++ti) acc[ti] = (f8){};

        for (int e0 = 0; e0 < E_; e0 += 32) {
            // B operand: lane = t, K = e (16 contiguous f32 -> f16)
            const f4* ap = (const f4*)(attnT + ((size_t)b * T_ + t) * E_ + e0 + (hib ? 16 : 0));
            f4 x0 = ap[0], x1 = ap[1], x2 = ap[2], x3 = ap[3];
            h16 Bv;
#pragma unroll
            for (int i = 0; i < 4; ++i) {
                Bv[i]      = (_Float16)x0[i];
                Bv[4 + i]  = (_Float16)x1[i];
                Bv[8 + i]  = (_Float16)x2[i];
                Bv[12 + i] = (_Float16)x3[i];
            }
#pragma unroll
            for (int ti = 0; ti < 4; ++ti) {
                const _Float16* wp = Woh + (size_t)(co0 + ti * 16 + ln) * E_ + e0 + (hib ? 16 : 0);
                h16 A = cat8(*(const h8*)wp, *(const h8*)(wp + 8));
                acc[ti] = WMMA_F16(A, Bv, acc[ti]);
            }
        }

#pragma unroll
        for (int ti = 0; ti < 4; ++ti) {
#pragma unroll
            for (int r = 0; r < 8; ++r) {
                const int co = co0 + ti * 16 + r + (hib ? 8 : 0);
                out[((size_t)b * C_ + co) * T_ + t] = acc[ti][r] + bo[co];
            }
        }
    }
}

// ---------------------------------------------------------------------------
extern "C" void kernel_launch(void* const* d_in, const int* in_sizes, int n_in,
                              void* d_out, int out_size, void* d_ws, size_t ws_size,
                              hipStream_t stream) {
    const float* x  = (const float*)d_in[0];
    const float* Wq = (const float*)d_in[1];
    const float* bq = (const float*)d_in[2];
    const float* Wk = (const float*)d_in[3];
    const float* bk = (const float*)d_in[4];
    const float* Wv = (const float*)d_in[5];
    const float* bv = (const float*)d_in[6];
    const float* Wp = (const float*)d_in[7];
    const float* bp = (const float*)d_in[8];
    const float* Wo = (const float*)d_in[9];
    const float* bo = (const float*)d_in[10];

    char* ws = (char*)d_ws;
    size_t off = 0;
    _Float16* whq = (_Float16*)(ws + off); off += (size_t)E_ * C_ * 2;          // 1 MB
    _Float16* whk = (_Float16*)(ws + off); off += (size_t)E_ * C_ * 2;
    _Float16* whv = (_Float16*)(ws + off); off += (size_t)E_ * C_ * 2;
    _Float16* whp = (_Float16*)(ws + off); off += (size_t)(H_ * P_) * C_ * 2;   // 128 KB
    _Float16* who = (_Float16*)(ws + off); off += (size_t)C_ * E_ * 2;          // 1 MB
    _Float16* qb  = (_Float16*)(ws + off); off += (size_t)B_ * T_ * E_ * 2;     // 2 MB
    _Float16* kb  = (_Float16*)(ws + off); off += (size_t)B_ * T_ * E_ * 2;
    _Float16* vb  = (_Float16*)(ws + off); off += (size_t)B_ * T_ * E_ * 2;
    _Float16* pb  = (_Float16*)(ws + off); off += (size_t)B_ * T_ * (H_*P_) * 2;
    float* slope  = (float*)(ws + off);    off += (size_t)B_ * H_ * T_ * 4;
    float* attnT  = (float*)(ws + off);    off += (size_t)B_ * T_ * E_ * 4;     // 4 MB

    // 1) weights -> f16
    const int nW = E_ * C_, nP = H_ * P_ * C_;
    cvt_f16_kernel<<<(nW + 255) / 256, 256, 0, stream>>>(Wq, whq, nW);
    cvt_f16_kernel<<<(nW + 255) / 256, 256, 0, stream>>>(Wk, whk, nW);
    cvt_f16_kernel<<<(nW + 255) / 256, 256, 0, stream>>>(Wv, whv, nW);
    cvt_f16_kernel<<<(nP + 255) / 256, 256, 0, stream>>>(Wp, whp, nP);
    cvt_f16_kernel<<<(nW + 255) / 256, 256, 0, stream>>>(Wo, who, nW);

    // 2) q/k/v/p projections (WMMA GEMMs)
    dim3 gq(T_ / 128, B_, E_ / 64);
    proj_gemm_kernel<4><<<gq, 256, 0, stream>>>(x, whq, bq, qb, E_);
    proj_gemm_kernel<4><<<gq, 256, 0, stream>>>(x, whk, bk, kb, E_);
    proj_gemm_kernel<4><<<gq, 256, 0, stream>>>(x, whv, bv, vb, E_);
    dim3 gp(T_ / 128, B_, 1);
    proj_gemm_kernel<2><<<gp, 256, 0, stream>>>(x, whp, bp, pb, H_ * P_);

    // 3) penalty slopes
    slope_kernel<<<(B_ * T_) / 256, 256, 0, stream>>>(pb, slope);

    // 4) fused flash attention
    dim3 ga(T_ / 128, B_ * H_);
    attn_kernel<<<ga, 256, 0, stream>>>(qb, kb, vb, slope, attnT);

    // 5) output projection
    dim3 go(T_ / 128, B_, C_ / 256);
    out_proj_kernel<<<go, 256, 0, stream>>>(attnT, who, bo, (float*)d_out);
}
